// ChromaDoubleStreamBlock_9732395893183
// MI455X (gfx1250) — matrix operand order repaired and checked by reading, not compiled, not run on hardware
//
#include <hip/hip_runtime.h>
#include <math.h>

// ---------------------------------------------------------------------------
// MI455X (gfx1250) implementation. Strategy: convert everything to bf16 once,
// run every matmul through v_wmma_f32_16x16x32_bf16 (wave32). fp32 weights are
// 870MB -> ~37us of mandatory HBM traffic at 23.3TB/s; 516 GFLOP of math needs
// the bf16 WMMA path to stay compute-bound. Each bf16 weight matrix (<=75.5MB)
// fits in the 192MB L2, so M-tile re-reads are L2 hits.
// GEMM tiles are double-buffered and staged with GLOBAL_LOAD_ASYNC_TO_LDS
// (ASYNCcnt) so the next tile's copy overlaps the current tile's WMMAs, one
// barrier per K-step instead of two, no VGPR staging round-trip.
// ---------------------------------------------------------------------------

typedef __bf16 bf16_t;
typedef bf16_t v16bf __attribute__((ext_vector_type(16)));
typedef bf16_t v8bf  __attribute__((ext_vector_type(8)));
typedef float  v8f   __attribute__((ext_vector_type(8)));
typedef int    v4i_t __attribute__((__vector_size__(16)));

#define DIMC   3072
#define HEADS  24
#define DHEAD  128
#define LTXT   512
#define LIMG   1536
#define SEQL   2048
#define MLPD   12288

#define BM   128
#define BN   128
#define BKK  64
#define LPAD 8   // pad LDS rows by 8 bf16 (16B) to break bank conflicts

#if __has_builtin(__builtin_amdgcn_global_load_async_to_lds_b128) && \
    __has_builtin(__builtin_amdgcn_s_wait_asynccnt)
#define HAS_ASYNC 1
#else
#define HAS_ASYNC 0
#endif

#if HAS_ASYNC
// One wave-wide async 16B/lane copy: global -> LDS, tracked by ASYNCcnt.
__device__ __forceinline__ void async16(const bf16_t* g, bf16_t* l) {
  __builtin_amdgcn_global_load_async_to_lds_b128(
      (__attribute__((address_space(1))) v4i_t*)(g),
      (__attribute__((address_space(3))) v4i_t*)(l), 0, 0);
}
#endif

// ---- WMMA fragment gathers (per CDNA5 ISA 7.12.2 VGPR layouts) -------------
// A (16x32, 16-bit): lane l holds row m=l&15; VGPR0..3 = K kh..kh+7,
// VGPR4..7 = K 16+kh..16+kh+7 with kh = 8*(l>>4).
__device__ __forceinline__ v16bf ld_fragA(const bf16_t* p) {
  v8bf lo = *(const v8bf*)p;
  v8bf hi = *(const v8bf*)(p + 16);
  v16bf f;
#pragma unroll
  for (int i = 0; i < 8; ++i) { f[i] = lo[i]; f[i + 8] = hi[i]; }
  return f;
}
// B (32x16, 16-bit): lane l holds column n=l&15; same K order as A:
// K kh..kh+7 then 16+kh..16+kh+7 with kh = 8*(l>>4).
__device__ __forceinline__ v16bf ld_fragB(const bf16_t* p) {
  v8bf lo = *(const v8bf*)p;
  v8bf hi = *(const v8bf*)(p + 16);
  v16bf f;
#pragma unroll
  for (int i = 0; i < 8; ++i) { f[i] = lo[i]; f[i + 8] = hi[i]; }
  return f;
}

// ---- Generic bf16 GEMM: C = A(MxK,row) * B(NxK,row)^T, f32 accum -----------
// EPI: 0 = +bias -> f32     (QKV / proj)
//      1 = *alpha, mask -> f32 (attention scores)
//      2 = +bias, gelu(tanh) -> bf16 (MLP up)
//      3 = +bias, resid + gate*x -> f32 (MLP down, final output)
//      4 = -> bf16 (attention P@V)
template <int EPI>
__global__ __launch_bounds__(256) void gemm_kernel(
    const bf16_t* __restrict__ A, const bf16_t* __restrict__ Bw,
    int M, int N, int K,
    long sAb, long sBb, long sCb,
    const float* __restrict__ bias, const float* __restrict__ gate,
    const float* __restrict__ resid, const unsigned char* __restrict__ mask,
    float* __restrict__ Cf, bf16_t* __restrict__ Cbf, int ldc, float alpha)
{
  __shared__ __attribute__((aligned(16))) bf16_t As[2][BM][BKK + LPAD];
  __shared__ __attribute__((aligned(16))) bf16_t Bs[2][BN][BKK + LPAD];
  (void)M;
  const int bz = blockIdx.z;
  A  += (size_t)bz * sAb;
  Bw += (size_t)bz * sBb;
  const size_t coff = (size_t)bz * sCb;
  const int m0 = blockIdx.y * BM, n0 = blockIdx.x * BN;
  const int tid = threadIdx.x, lane = tid & 31, wave = tid >> 5;
  const int wm = (wave >> 1) * 32;   // 4 waves along M
  const int wn = (wave & 1) * 64;    // 2 waves along N

  v8f acc[2][4];
#pragma unroll
  for (int i = 0; i < 2; ++i)
#pragma unroll
    for (int j = 0; j < 4; ++j) {
      v8f z = {0.f, 0.f, 0.f, 0.f, 0.f, 0.f, 0.f, 0.f};
      acc[i][j] = z;
    }

  const int r  = lane & 15;
  const int kh = (lane >> 4) << 3;   // A fragment K-half
  const int kb = (lane >> 4) << 4;   // unused; B fragments also use kh

#if HAS_ASYNC
  auto load_tile = [&](int buf, int kk) {
#pragma unroll
    for (int c = 0; c < 4; ++c) {           // 128x64 bf16 tile: 32 elems/thread
      int q   = tid + c * 256;
      int row = q >> 3;
      int col = (q & 7) << 3;
      async16(&A [(size_t)(m0 + row) * K + kk + col], &As[buf][row][col]);
      async16(&Bw[(size_t)(n0 + row) * K + kk + col], &Bs[buf][row][col]);
    }
  };
  int buf = 0;
  load_tile(0, 0);
  for (int kk = 0; kk < K; kk += BKK) {
    __builtin_amdgcn_s_wait_asynccnt(0);   // this wave's tile landed in LDS
    __syncthreads();                       // everyone's landed; prev compute done
    if (kk + BKK < K) load_tile(buf ^ 1, kk + BKK);  // overlap next copy
#pragma unroll
    for (int ks = 0; ks < BKK; ks += 32) {
      v16bf a[2], b[4];
#pragma unroll
      for (int i = 0; i < 2; ++i) a[i] = ld_fragA(&As[buf][wm + i * 16 + r][ks + kh]);
#pragma unroll
      for (int j = 0; j < 4; ++j) b[j] = ld_fragB(&Bs[buf][wn + j * 16 + r][ks + kh]);
#pragma unroll
      for (int i = 0; i < 2; ++i)
#pragma unroll
        for (int j = 0; j < 4; ++j)
          acc[i][j] = __builtin_amdgcn_wmma_f32_16x16x32_bf16(
              false, a[i], false, b[j], (short)0, acc[i][j], false, false);
    }
    buf ^= 1;
  }
#else
  for (int kk = 0; kk < K; kk += BKK) {
#pragma unroll
    for (int c = 0; c < 4; ++c) {
      int q   = tid + c * 256;
      int row = q >> 3;
      int col = (q & 7) << 3;
      *(v8bf*)&As[0][row][col] = *(const v8bf*)&A [(size_t)(m0 + row) * K + kk + col];
      *(v8bf*)&Bs[0][row][col] = *(const v8bf*)&Bw[(size_t)(n0 + row) * K + kk + col];
    }
    __syncthreads();
#pragma unroll
    for (int ks = 0; ks < BKK; ks += 32) {
      v16bf a[2], b[4];
#pragma unroll
      for (int i = 0; i < 2; ++i) a[i] = ld_fragA(&As[0][wm + i * 16 + r][ks + kh]);
#pragma unroll
      for (int j = 0; j < 4; ++j) b[j] = ld_fragB(&Bs[0][wn + j * 16 + r][ks + kh]);
#pragma unroll
      for (int i = 0; i < 2; ++i)
#pragma unroll
        for (int j = 0; j < 4; ++j)
          acc[i][j] = __builtin_amdgcn_wmma_f32_16x16x32_bf16(
              false, a[i], false, b[j], (short)0, acc[i][j], false, false);
    }
    __syncthreads();
  }
#endif

  // C/D layout: lane -> (n = lane&15, m = 8*(lane>>4) + vgpr)
  const int mh = (lane >> 4) << 3;
#pragma unroll
  for (int i = 0; i < 2; ++i)
#pragma unroll
    for (int j = 0; j < 4; ++j) {
      const int gn = n0 + wn + j * 16 + r;
      const float bv = bias ? bias[gn] : 0.f;
#pragma unroll
      for (int v = 0; v < 8; ++v) {
        const int gm = m0 + wm + i * 16 + mh + v;
        float x = acc[i][j][v] * alpha + bv;
        const size_t ci = coff + (size_t)gm * ldc + gn;
        if (EPI == 0) {
          Cf[ci] = x;
        } else if (EPI == 1) {
          if (mask && !mask[(size_t)gm * N + gn]) x = -__builtin_inff();
          Cf[ci] = x;
        } else if (EPI == 2) {
          float t = 0.7978845608028654f * (x + 0.044715f * x * x * x);
          Cbf[ci] = (bf16_t)(0.5f * x * (1.f + tanhf(t)));
        } else if (EPI == 3) {
          Cf[ci] = resid[(size_t)gm * ldc + gn] + gate[gn] * x;
        } else {
          Cbf[ci] = (bf16_t)x;
        }
      }
    }
}

// ---- fp32 -> bf16 conversion ----------------------------------------------
__global__ void cvt_kernel(const float* __restrict__ in, bf16_t* __restrict__ out, long n) {
  long i = ((long)blockIdx.x * blockDim.x + threadIdx.x) * 4;
  if (i + 3 < n) {
    float4 v = *(const float4*)(in + i);
    out[i]     = (bf16_t)v.x;
    out[i + 1] = (bf16_t)v.y;
    out[i + 2] = (bf16_t)v.z;
    out[i + 3] = (bf16_t)v.w;
  } else {
    for (; i < n; ++i) out[i] = (bf16_t)in[i];
  }
}

// ---- LayerNorm + (1+scale)*x + shift -> bf16 ------------------------------
__global__ __launch_bounds__(256) void ln_mod_kernel(
    const float* __restrict__ x, const float* __restrict__ scale,
    const float* __restrict__ shift, bf16_t* __restrict__ out) {
  const size_t row = blockIdx.x;
  const float* xr = x + row * DIMC;
  bf16_t* orow = out + row * DIMC;
  __shared__ float s1[256], s2[256];
  const int t = threadIdx.x;
  float a = 0.f, b = 0.f;
  for (int i = t; i < DIMC; i += 256) { float v = xr[i]; a += v; b += v * v; }
  s1[t] = a; s2[t] = b; __syncthreads();
  for (int o = 128; o > 0; o >>= 1) {
    if (t < o) { s1[t] += s1[t + o]; s2[t] += s2[t + o]; }
    __syncthreads();
  }
  const float mean = s1[0] * (1.f / DIMC);
  const float var  = s2[0] * (1.f / DIMC) - mean * mean;
  const float inv  = rsqrtf(var + 1e-6f);
  for (int i = t; i < DIMC; i += 256)
    orow[i] = (bf16_t)((1.f + scale[i]) * ((xr[i] - mean) * inv) + shift[i]);
}

// ---- QKV split + RMS qk-norm + RoPE; V stored transposed [h][dh][l] -------
__global__ __launch_bounds__(128) void qkvpost_kernel(
    const float* __restrict__ qkv, const float* __restrict__ pe,
    const float* __restrict__ txt_w, const float* __restrict__ img_w,
    bf16_t* __restrict__ Qb, bf16_t* __restrict__ Kb, bf16_t* __restrict__ Vt) {
  const int l = blockIdx.x, h = blockIdx.y, d = threadIdx.x;
  const float* rowp = qkv + (size_t)l * (3 * DIMC) + h * DHEAD;
  const float q = rowp[d], k = rowp[DIMC + d], v = rowp[2 * DIMC + d];
  const float* w = (l < LTXT) ? txt_w : img_w;
  __shared__ float red[128];
  __shared__ float sq[128], sk[128];
  red[d] = q * q; __syncthreads();
  for (int o = 64; o > 0; o >>= 1) { if (d < o) red[d] += red[d + o]; __syncthreads(); }
  const float qinv = rsqrtf(red[0] * (1.f / DHEAD) + 1e-6f);
  __syncthreads();
  red[d] = k * k; __syncthreads();
  for (int o = 64; o > 0; o >>= 1) { if (d < o) red[d] += red[d + o]; __syncthreads(); }
  const float kinv = rsqrtf(red[0] * (1.f / DHEAD) + 1e-6f);
  sq[d] = q * qinv * w[d];
  sk[d] = k * kinv * w[d];
  __syncthreads();
  // pe_freqs_cis (1,1,L,64,2,2): (l,j,a,b) at l*256 + j*4 + a*2 + b
  const int j = d >> 1;
  const float* pr = pe + (size_t)l * 256 + j * 4;
  const float c = pr[0], s = pr[2];
  float qo, ko;
  if ((d & 1) == 0) { qo = sq[d] * c - sq[d + 1] * s; ko = sk[d] * c - sk[d + 1] * s; }
  else              { qo = sq[d - 1] * s + sq[d] * c; ko = sk[d - 1] * s + sk[d] * c; }
  const size_t qi = ((size_t)h * SEQL + l) * DHEAD + d;
  Qb[qi] = (bf16_t)qo;
  Kb[qi] = (bf16_t)ko;
  Vt[((size_t)h * DHEAD + d) * SEQL + l] = (bf16_t)v;   // [h][dh][l] for B-frag
}

// ---- Row softmax over 2048 keys -> bf16 probabilities ---------------------
__global__ __launch_bounds__(256) void softmax_kernel(
    const float* __restrict__ S, bf16_t* __restrict__ P) {
  const size_t row = blockIdx.x;
  const float* s = S + row * SEQL;
  bf16_t* p = P + row * SEQL;
  __shared__ float red[256];
  const int t = threadIdx.x;
  float vals[8];
  float m = -3.4028235e38f;
#pragma unroll
  for (int i = 0; i < 8; ++i) { vals[i] = s[t + i * 256]; m = fmaxf(m, vals[i]); }
  red[t] = m; __syncthreads();
  for (int o = 128; o > 0; o >>= 1) { if (t < o) red[t] = fmaxf(red[t], red[t + o]); __syncthreads(); }
  m = red[0]; __syncthreads();
  float sum = 0.f;
#pragma unroll
  for (int i = 0; i < 8; ++i) { vals[i] = __expf(vals[i] - m); sum += vals[i]; }
  red[t] = sum; __syncthreads();
  for (int o = 128; o > 0; o >>= 1) { if (t < o) red[t] += red[t + o]; __syncthreads(); }
  const float inv = 1.f / red[0];
#pragma unroll
  for (int i = 0; i < 8; ++i) p[t + i * 256] = (bf16_t)(vals[i] * inv);
}

// ---- resid = embeds + gate * proj_out -------------------------------------
__global__ void resid_kernel(const float* __restrict__ emb, const float* __restrict__ gate,
                             const float* __restrict__ proj, float* __restrict__ res, long n) {
  long i = (long)blockIdx.x * blockDim.x + threadIdx.x;
  if (i >= n) return;
  int col = (int)(i % DIMC);
  res[i] = emb[i] + gate[col] * proj[i];
}

// ---------------------------------------------------------------------------
extern "C" void kernel_launch(void* const* d_in, const int* in_sizes, int n_in,
                              void* d_out, int out_size, void* d_ws, size_t ws_size,
                              hipStream_t stream) {
  (void)in_sizes; (void)n_in; (void)out_size; (void)ws_size;
  const float* img_embeds     = (const float*)d_in[0];
  const float* txt_embeds     = (const float*)d_in[1];
  const float* pe             = (const float*)d_in[2];
  const float* img_attn_scale = (const float*)d_in[3];
  const float* img_attn_shift = (const float*)d_in[4];
  const float* img_attn_gate  = (const float*)d_in[5];
  const float* img_mlp_scale  = (const float*)d_in[6];
  const float* img_mlp_shift  = (const float*)d_in[7];
  const float* img_mlp_gate   = (const float*)d_in[8];
  const float* txt_attn_scale = (const float*)d_in[9];
  const float* txt_attn_shift = (const float*)d_in[10];
  const float* txt_attn_gate  = (const float*)d_in[11];
  const float* txt_mlp_scale  = (const float*)d_in[12];
  const float* txt_mlp_shift  = (const float*)d_in[13];
  const float* txt_mlp_gate   = (const float*)d_in[14];
  const float* img_qkv_w      = (const float*)d_in[15];
  const float* img_qkv_b      = (const float*)d_in[16];
  const float* img_qknorm_w   = (const float*)d_in[17];
  const float* txt_qkv_w      = (const float*)d_in[18];
  const float* txt_qkv_b      = (const float*)d_in[19];
  const float* txt_qknorm_w   = (const float*)d_in[20];
  const float* proj_w         = (const float*)d_in[21];
  const float* proj_b         = (const float*)d_in[22];
  const float* img_mlp_w1     = (const float*)d_in[23];
  const float* img_mlp_b1     = (const float*)d_in[24];
  const float* img_mlp_w2     = (const float*)d_in[25];
  const float* img_mlp_b2     = (const float*)d_in[26];
  const float* txt_mlp_w1     = (const float*)d_in[27];
  const float* txt_mlp_b1     = (const float*)d_in[28];
  const float* txt_mlp_w2     = (const float*)d_in[29];
  const float* txt_mlp_b2     = (const float*)d_in[30];
  const unsigned char* mask   = (const unsigned char*)d_in[31];
  float* out = (float*)d_out;

  // ---- workspace layout ----
  char* base = (char*)d_ws;
  size_t off = 0;
  auto alloc = [&](size_t bytes) -> void* {
    void* p = base + off;
    off += (bytes + 255) & ~(size_t)255;
    return p;
  };
  bf16_t* Wqkv_t = (bf16_t*)alloc((size_t)3 * DIMC * DIMC * 2);
  bf16_t* Wqkv_i = (bf16_t*)alloc((size_t)3 * DIMC * DIMC * 2);
  bf16_t* Wproj  = (bf16_t*)alloc((size_t)DIMC * DIMC * 2);
  bf16_t* Wm1i   = (bf16_t*)alloc((size_t)MLPD * DIMC * 2);
  bf16_t* Wm2i   = (bf16_t*)alloc((size_t)DIMC * MLPD * 2);
  bf16_t* Wm1t   = (bf16_t*)alloc((size_t)MLPD * DIMC * 2);
  bf16_t* Wm2t   = (bf16_t*)alloc((size_t)DIMC * MLPD * 2);
  bf16_t* modbuf = (bf16_t*)alloc((size_t)SEQL * DIMC * 2);
  float*  qkvbuf = (float*) alloc((size_t)SEQL * 3 * DIMC * 4);
  bf16_t* Qb     = (bf16_t*)alloc((size_t)HEADS * SEQL * DHEAD * 2);
  bf16_t* Kb     = (bf16_t*)alloc((size_t)HEADS * SEQL * DHEAD * 2);
  bf16_t* Vt     = (bf16_t*)alloc((size_t)HEADS * SEQL * DHEAD * 2);
  float*  scores = (float*) alloc((size_t)HEADS * SEQL * SEQL * 4);
  bf16_t* Pbuf   = (bf16_t*)alloc((size_t)HEADS * SEQL * SEQL * 2);
  bf16_t* Obuf   = (bf16_t*)alloc((size_t)SEQL * DIMC * 2);
  float*  projout= (float*) alloc((size_t)SEQL * DIMC * 4);
  float*  residb = (float*) alloc((size_t)SEQL * DIMC * 4);
  bf16_t* mlpin  = (bf16_t*)alloc((size_t)SEQL * DIMC * 2);
  bf16_t* hidden = (bf16_t*)alloc((size_t)LIMG * MLPD * 2);

  auto cvt = [&](const float* src, bf16_t* dst, long n) {
    cvt_kernel<<<dim3((unsigned)((n / 4 + 255) / 256)), 256, 0, stream>>>(src, dst, n);
  };

  // 1) weights -> bf16
  cvt(txt_qkv_w, Wqkv_t, 3L * DIMC * DIMC);
  cvt(img_qkv_w, Wqkv_i, 3L * DIMC * DIMC);
  cvt(proj_w,    Wproj,  (long)DIMC * DIMC);
  cvt(img_mlp_w1, Wm1i, (long)MLPD * DIMC);
  cvt(img_mlp_w2, Wm2i, (long)DIMC * MLPD);
  cvt(txt_mlp_w1, Wm1t, (long)MLPD * DIMC);
  cvt(txt_mlp_w2, Wm2t, (long)DIMC * MLPD);

  // 2) LN + attention modulation (txt rows first, matching concat order)
  ln_mod_kernel<<<LTXT, 256, 0, stream>>>(txt_embeds, txt_attn_scale, txt_attn_shift, modbuf);
  ln_mod_kernel<<<LIMG, 256, 0, stream>>>(img_embeds, img_attn_scale, img_attn_shift,
                                          modbuf + (size_t)LTXT * DIMC);

  // 3) QKV GEMMs (separate weights per stream), fp32 out + bias
  gemm_kernel<0><<<dim3(3 * DIMC / BN, LTXT / BM, 1), 256, 0, stream>>>(
      modbuf, Wqkv_t, LTXT, 3 * DIMC, DIMC, 0, 0, 0,
      txt_qkv_b, nullptr, nullptr, nullptr, qkvbuf, nullptr, 3 * DIMC, 1.f);
  gemm_kernel<0><<<dim3(3 * DIMC / BN, LIMG / BM, 1), 256, 0, stream>>>(
      modbuf + (size_t)LTXT * DIMC, Wqkv_i, LIMG, 3 * DIMC, DIMC, 0, 0, 0,
      img_qkv_b, nullptr, nullptr, nullptr,
      qkvbuf + (size_t)LTXT * 3 * DIMC, nullptr, 3 * DIMC, 1.f);

  // 4) RMS qk-norm + RoPE; V transposed for the P@V B-operand
  qkvpost_kernel<<<dim3(SEQL, HEADS), 128, 0, stream>>>(
      qkvbuf, pe, txt_qknorm_w, img_qknorm_w, Qb, Kb, Vt);

  // 5) scores = (Q K^T) * DH^-0.5, masked; batched over heads
  gemm_kernel<1><<<dim3(SEQL / BN, SEQL / BM, HEADS), 256, 0, stream>>>(
      Qb, Kb, SEQL, SEQL, DHEAD,
      (long)SEQL * DHEAD, (long)SEQL * DHEAD, (long)SEQL * SEQL,
      nullptr, nullptr, nullptr, mask, scores, nullptr, SEQL, 0.08838834764831845f);

  // 6) softmax rows -> bf16 P
  softmax_kernel<<<HEADS * SEQL, 256, 0, stream>>>(scores, Pbuf);

  // 7) O = P @ V, written interleaved as [l][h*128+dh]
  gemm_kernel<4><<<dim3(1, SEQL / BM, HEADS), 256, 0, stream>>>(
      Pbuf, Vt, SEQL, DHEAD, SEQL,
      (long)SEQL * SEQL, (long)DHEAD * SEQL, (long)DHEAD,
      nullptr, nullptr, nullptr, nullptr, nullptr, Obuf, DIMC, 1.f);

  // 8) shared output projection
  gemm_kernel<0><<<dim3(DIMC / BN, SEQL / BM, 1), 256, 0, stream>>>(
      Obuf, Wproj, SEQL, DIMC, DIMC, 0, 0, 0,
      proj_b, nullptr, nullptr, nullptr, projout, nullptr, DIMC, 1.f);

  // 9) residual + attention gate
  {
    long nt = (long)LTXT * DIMC, ni = (long)LIMG * DIMC;
    resid_kernel<<<dim3((unsigned)((nt + 255) / 256)), 256, 0, stream>>>(
        txt_embeds, txt_attn_gate, projout, residb, nt);
    resid_kernel<<<dim3((unsigned)((ni + 255) / 256)), 256, 0, stream>>>(
        img_embeds, img_attn_gate, projout + (size_t)LTXT * DIMC,
        residb + (size_t)LTXT * DIMC, ni);
  }

  // 10) LN + MLP modulation
  ln_mod_kernel<<<LTXT, 256, 0, stream>>>(residb, txt_mlp_scale, txt_mlp_shift, mlpin);
  ln_mod_kernel<<<LIMG, 256, 0, stream>>>(residb + (size_t)LTXT * DIMC,
                                          img_mlp_scale, img_mlp_shift,
                                          mlpin + (size_t)LTXT * DIMC);

  // 11) img MLP: up (bias+gelu->bf16), down (bias + resid + gate -> d_out img)
  gemm_kernel<2><<<dim3(MLPD / BN, LIMG / BM, 1), 256, 0, stream>>>(
      mlpin + (size_t)LTXT * DIMC, Wm1i, LIMG, MLPD, DIMC, 0, 0, 0,
      img_mlp_b1, nullptr, nullptr, nullptr, nullptr, hidden, MLPD, 1.f);
  gemm_kernel<3><<<dim3(DIMC / BN, LIMG / BM, 1), 256, 0, stream>>>(
      hidden, Wm2i, LIMG, DIMC, MLPD, 0, 0, 0,
      img_mlp_b2, img_mlp_gate, residb + (size_t)LTXT * DIMC, nullptr,
      out, nullptr, DIMC, 1.f);

  // 12) txt MLP (reuse hidden buffer) -> d_out txt region
  gemm_kernel<2><<<dim3(MLPD / BN, LTXT / BM, 1), 256, 0, stream>>>(
      mlpin, Wm1t, LTXT, MLPD, DIMC, 0, 0, 0,
      txt_mlp_b1, nullptr, nullptr, nullptr, nullptr, hidden, MLPD, 1.f);
  gemm_kernel<3><<<dim3(DIMC / BN, LTXT / BM, 1), 256, 0, stream>>>(
      hidden, Wm2t, LTXT, DIMC, MLPD, 0, 0, 0,
      txt_mlp_b2, txt_mlp_gate, residb, nullptr,
      out + (size_t)LIMG * DIMC, nullptr, DIMC, 1.f);
}
